// DenoisingDecoder_18932215841128
// MI455X (gfx1250) — matrix operand order repaired and checked by reading, not compiled, not run on hardware
//
#include <hip/hip_runtime.h>
#include <hip/hip_fp16.h>
#include <hip/hip_bf16.h>

// ---------------------------------------------------------------------------
// DenoisingDecoder for MI455X (gfx1250, wave32, WMMA + TDM)
// Dominant cost: per-layer edge GEMM (B*N*N, 288) x (288, 256) fused with
// sigmoid/silu gating, pair mask and j-reduction -> WMMA f32_16x16x32_f16.
// LDS staging of the per-batch h tile is done by the Tensor Data Mover with
// pad_enable producing a conflict-free 272B row stride.
// ---------------------------------------------------------------------------

typedef __attribute__((ext_vector_type(16))) _Float16 v16h;
typedef __attribute__((ext_vector_type(8)))  float    v8f;
typedef __attribute__((ext_vector_type(4)))  unsigned int u32x4;
typedef __attribute__((ext_vector_type(8)))  int      i32x8;
typedef __attribute__((ext_vector_type(4)))  int      i32x4;

union AF { uint4 u[2]; v16h v; };

#if defined(__has_builtin)
#if __has_builtin(__builtin_amdgcn_tensor_load_to_lds)
#define HAVE_TDM 1
#else
#define HAVE_TDM 0
#endif
#else
#define HAVE_TDM 0
#endif

constexpr int NB   = 8;            // batch
constexpr int NA   = 128;          // atoms
constexpr int HD   = 128;          // hidden dim
constexpr int NL   = 3;            // layers
constexpr int RBIN = 32;           // rbf bins
constexpr int KD   = 2*HD + RBIN;  // 288 = GEMM K
constexpr int KS   = KD / 32;      // 9 wmma k-steps
constexpr int VOC  = 100;

// workspace byte offsets (all 256-aligned; total ~10.5 MB)
constexpr size_t OFF_CART = 0;                    // (B,N,3) f32       12288
constexpr size_t OFF_COND = 12288;                // (B,256) f32        8192
constexpr size_t OFF_CND  = 20480;                // (B,128) f32        4096
constexpr size_t OFF_WSWZ = 24576;                // swizzled edge_W f16 442368
constexpr size_t OFF_H16A = 466944;               // (B,N,H) f16      262144
constexpr size_t OFF_H16B = 729088;               // (B,N,H) f16      262144
constexpr size_t OFF_HFA  = 991232;               // (B,N,H) f32      524288
constexpr size_t OFF_HFB  = 1515520;              // (B,N,H) f32      524288
constexpr size_t OFF_TMP  = 2039808;              // (B*N,H) f32      524288
constexpr size_t OFF_RBF  = 2564096;              // (B,N,N,32) f16  8388608

__device__ __forceinline__ float sigm(float x) { return 1.f / (1.f + __expf(-x)); }

#if HAVE_TDM
// TDM 2D tile load Global->LDS with row padding.
// rowElems elements of 2 bytes per row, `rows` rows, global row stride
// strideElems elements; LDS gets padAmountDW dwords of padding appended after
// every padIntervalCode-encoded dword interval (code: 0=2,1=4,...,5=64 DW).
__device__ __forceinline__ void tdm_load_2d_to_lds(const void* gptr, unsigned lds_off,
                                                   unsigned rowElems, unsigned rows,
                                                   unsigned strideElems,
                                                   unsigned padIntervalCode,
                                                   unsigned padAmountCode,
                                                   unsigned padEnable) {
  unsigned long long ga = (unsigned long long)gptr;
  u32x4 g0;
  g0[0] = 1u;                                          // count=1, user D#
  g0[1] = lds_off;                                     // lds_addr (bytes)
  g0[2] = (unsigned)(ga & 0xFFFFFFFFu);                // global_addr[31:0]
  g0[3] = (unsigned)((ga >> 32) & 0x1FFFFFFu)          // global_addr[56:32]
        | (2u << 30);                                  // type=2 (image)
  i32x8 g1;
  g1[0] = (int)((1u << 16)                             // data_size=1 -> 2B
        | (padEnable << 20)
        | (padIntervalCode << 22)
        | (padAmountCode << 25));
  g1[1] = (int)(rowElems << 16);                       // tensor_dim0[15:0]
  g1[2] = (int)((rowElems >> 16) | (rows << 16));      // dim0 hi | tensor_dim1 lo
  g1[3] = (int)((rows >> 16) | (rowElems << 16));      // dim1 hi | tile_dim0
  g1[4] = (int)rows;                                   // tile_dim1 (tile_dim2=0)
  g1[5] = (int)strideElems;                            // tensor_dim0_stride lo
  g1[6] = 0;                                           // stride hi | dim1_stride lo
  g1[7] = 0;
  i32x4 gz = {0, 0, 0, 0};
#if __clang_major__ >= 23
  i32x8 gz8 = {0, 0, 0, 0, 0, 0, 0, 0};
  __builtin_amdgcn_tensor_load_to_lds(g0, g1, gz, gz, gz8, 0);
#else
  __builtin_amdgcn_tensor_load_to_lds(g0, g1, gz, gz, 0);
#endif
  __builtin_amdgcn_s_wait_tensorcnt(0);
}
#endif

// ------------------------- setup kernels -----------------------------------

__global__ void dd_cart_kernel(const float* __restrict__ frac,
                               const float* __restrict__ lat,
                               float* __restrict__ cart) {
  int idx = blockIdx.x * blockDim.x + threadIdx.x;       // b*NA + n
  if (idx >= NB * NA) return;
  int b = idx >> 7;
  const float* f = frac + idx * 3;
  const float* L = lat + b * 9;
  #pragma unroll
  for (int j = 0; j < 3; ++j)
    cart[idx * 3 + j] = f[0] * L[0 * 3 + j] + f[1] * L[1 * 3 + j] + f[2] * L[2 * 3 + j];
}

__global__ void dd_rbf_kernel(const float* __restrict__ cart,
                              _Float16* __restrict__ rbf) {
  int idx = blockIdx.x * blockDim.x + threadIdx.x;       // b*NA*NA + n*NA + m
  if (idx >= NB * NA * NA) return;
  int mcol = idx & 127;
  int rest = idx >> 7;
  int nrow = rest & 127;
  int b    = rest >> 7;
  const float* cn = cart + ((size_t)b * NA + nrow) * 3;
  const float* cm = cart + ((size_t)b * NA + mcol) * 3;
  float dx = cn[0] - cm[0], dy = cn[1] - cm[1], dz = cn[2] - cm[2];
  float dist = sqrtf(dx * dx + dy * dy + dz * dz + 1e-6f);
  const float step  = 8.0f / 31.0f;
  const float gamma = 1.0f / (step * step);
  _Float16* out = rbf + (size_t)idx * RBIN;
  #pragma unroll 8
  for (int k = 0; k < RBIN; ++k) {
    float d = dist - step * (float)k;
    out[k] = (_Float16)__expf(-gamma * d * d);
  }
}

__global__ void dd_h0_kernel(const int* __restrict__ types,
                             const float* __restrict__ emb,
                             float* __restrict__ hf, _Float16* __restrict__ h16) {
  int idx = blockIdx.x * blockDim.x + threadIdx.x;       // (b*NA+n)*HD + c
  if (idx >= NB * NA * HD) return;
  int row = idx >> 7, c = idx & 127;
  float v = emb[types[row] * HD + c];
  hf[idx]  = v;
  h16[idx] = (_Float16)v;
}

// Pre-swizzle edge_W (L,288,256) f32 into WMMA B-fragment lane order (f16):
// frag (l,s,t) -> 32 lanes x 16 halves contiguous. Lane: n=lane&15,
// kbase=(lane>>4)*16, half p -> W[l, s*32+(p>>3)*16+kbase/2+(p&7), t*16+n].
__global__ void dd_wswz_kernel(const float* __restrict__ W,
                               _Float16* __restrict__ wz) {
  int idx = blockIdx.x * blockDim.x + threadIdx.x;       // ((l*9+s)*16+t)*32+lane
  if (idx >= NL * KS * 16 * 32) return;
  int lane = idx & 31;
  int r1   = idx >> 5;
  int t    = r1 & 15;
  int r2   = r1 >> 4;
  int s    = r2 % KS;
  int l    = r2 / KS;
  int kbase = (lane >> 4) * 16;
  int col   = t * 16 + (lane & 15);
  _Float16* dst = wz + (size_t)idx * 16;
  #pragma unroll
  for (int p = 0; p < 16; ++p) {
    int krow = s * 32 + (p & 8) * 2 + (kbase >> 1) + (p & 7);
    dst[p] = (_Float16)W[((size_t)l * KD + krow) * 256 + col];
  }
}

__global__ void dd_cond_kernel(const int* __restrict__ tstep,
                               const float* __restrict__ z,
                               const float* __restrict__ Wlat,
                               const float* __restrict__ blat,
                               float* __restrict__ cond) {
  int idx = blockIdx.x * blockDim.x + threadIdx.x;       // b*256 + c
  if (idx >= NB * 256) return;
  int b = idx >> 8, c = idx & 255;
  float v;
  if (c < 128) {
    float tf = (float)tstep[b];
    const float nlog = -9.210340371976184f;              // -ln(10000)
    if (c < 64) {
      float fr = __expf(nlog * (float)c / 63.0f);
      v = __sinf(tf * fr);
    } else {
      float fr = __expf(nlog * (float)(c - 64) / 63.0f);
      v = __cosf(tf * fr);
    }
  } else {
    int cc = c - 128;
    float acc = blat[cc];
    #pragma unroll 8
    for (int k = 0; k < 64; ++k) acc += z[b * 64 + k] * Wlat[k * HD + cc];
    v = acc;
  }
  cond[idx] = v;
}

__global__ void dd_cnd_kernel(const float* __restrict__ cond,
                              const float* __restrict__ cW,   // (256,128) layer slice
                              const float* __restrict__ cb,   // (128,)   layer slice
                              float* __restrict__ cnd) {
  int idx = blockIdx.x * blockDim.x + threadIdx.x;       // b*128 + c
  if (idx >= NB * HD) return;
  int b = idx >> 7, c = idx & 127;
  float acc = cb[c];
  #pragma unroll 8
  for (int f = 0; f < 256; ++f) acc += cond[b * 256 + f] * cW[f * HD + c];
  cnd[idx] = acc * sigm(acc);                            // silu
}

// ------------------------- fused edge-message layer -------------------------
// grid: (NA, NB); block: 256 = 8 waves. Wave w handles j-tile w for atom i.
// A rows (M) = 16 j's; K sections: [h_i broadcast | h_j | rbf_ij]; N-tile pair
// (t gate, t+8 core) kept in two f32 accumulators; gate/silu/mask applied in
// registers, reduced over M, cross-wave reduced in LDS.
__global__ __launch_bounds__(256) void dd_edge_kernel(
    const _Float16* __restrict__ h16_in,   // (B,N,H) f16
    const float*    __restrict__ hf_in,    // (B,N,H) f32
    const _Float16* __restrict__ rbf16,    // (B,N,N,32) f16
    const uint4*    __restrict__ wz,       // swizzled W frags, layer base
    const float*    __restrict__ ebias,    // (256,) layer slice
    const float*    __restrict__ cnd,      // (B,128)
    const unsigned char* __restrict__ mask,// (B,N)
    float*          __restrict__ hf_out,
    _Float16*       __restrict__ h16_out) {
  __shared__ _Float16 sh[NA * 136];        // padded h tile, conflict-free b128
  __shared__ float    smask[NA];
  __shared__ float    sagg[8 * HD];
  const int b    = blockIdx.y;
  const int i    = blockIdx.x;
  const int wv   = threadIdx.x >> 5;
  const int lane = threadIdx.x & 31;

  // stage h16[b,:,:] into LDS (row stride 136 halves = 272B)
#if HAVE_TDM
  if (wv == 0) {
    // TDM: 128 rows x 128 halves; pad 4 DWORDs after every 64 DWORDs (one
    // row) -> LDS row stride 272B. pad_interval code 5 = 64 DW, amount code
    // 3 = 4 DW.
    tdm_load_2d_to_lds(h16_in + (size_t)b * NA * HD,
                       (unsigned)(unsigned long long)(void*)&sh[0],
                       NA, NA, NA, 5u, 3u, 1u);
  }
#else
  {
    const uint4* src = (const uint4*)(h16_in + (size_t)b * NA * HD);
    for (int idx = threadIdx.x; idx < NA * 16; idx += 256) {
      int r = idx >> 4, c = idx & 15;
      *(uint4*)(&sh[r * 136 + c * 8]) = src[idx];
    }
  }
#endif
  for (int idx = threadIdx.x; idx < NA; idx += 256)
    smask[idx] = mask[b * NA + idx] ? 1.0f : 0.0f;
  __syncthreads();

  const int jt    = wv;
  const int j0    = jt * 16;
  const int m     = lane & 15;             // M row within tile / N col
  const int hi    = lane >> 4;
  const int koff8 = hi * 8;                // half offset within 32-K section

  // rbf A fragment for this (i, j-tile): rows j0+m, 32 K halves
  AF arbf;
  {
    const uint4* rp = (const uint4*)(rbf16 + ((size_t)(b * NA + i) * NA + j0 + m) * RBIN);
    arbf.u[0] = rp[hi];                    // halves [koff8 .. koff8+7]
    arbf.u[1] = rp[2 + hi];                // halves [16+koff8 .. 16+koff8+7]
  }
  float mj[8];
  #pragma unroll
  for (int r = 0; r < 8; ++r) mj[r] = smask[j0 + r + (hi << 3)];

  const _Float16* rowi = &sh[i * 136];
  const _Float16* rowj = &sh[(j0 + m) * 136];

  for (int t = 0; t < 8; ++t) {
    // prefetch next t's gate/core weight fragments (global_prefetch_b8)
    if (t < 7) {
      __builtin_prefetch(wz + (size_t)((t + 1) * 32 + lane) * 2, 0, 0);
      __builtin_prefetch(wz + (size_t)((t + 9) * 32 + lane) * 2, 0, 0);
    }
    float bg = ebias[t * 16 + m];
    float bc = ebias[128 + t * 16 + m];
    v8f accg = {bg, bg, bg, bg, bg, bg, bg, bg};
    v8f accc = {bc, bc, bc, bc, bc, bc, bc, bc};
    #pragma unroll
    for (int s = 0; s < KS; ++s) {
      AF a;
      if (s < 4) {                         // h_i: broadcast row i
        a.u[0] = *(const uint4*)(rowi + s * 32 + koff8);
        a.u[1] = *(const uint4*)(rowi + s * 32 + 16 + koff8);
      } else if (s < 8) {                  // h_j: row j0+m
        a.u[0] = *(const uint4*)(rowj + (s - 4) * 32 + koff8);
        a.u[1] = *(const uint4*)(rowj + (s - 4) * 32 + 16 + koff8);
      } else {                             // rbf section
        a = arbf;
      }
      AF bgf, bcf;
      const uint4* wg = wz + (size_t)((s * 16 + t) * 32 + lane) * 2;
      bgf.u[0] = wg[0]; bgf.u[1] = wg[1];
      const uint4* wc = wz + (size_t)((s * 16 + t + 8) * 32 + lane) * 2;
      bcf.u[0] = wc[0]; bcf.u[1] = wc[1];
      accg = __builtin_amdgcn_wmma_f32_16x16x32_f16(false, a.v, false, bgf.v,
                                                    (short)0, accg, false, false);
      accc = __builtin_amdgcn_wmma_f32_16x16x32_f16(false, a.v, false, bcf.v,
                                                    (short)0, accc, false, false);
    }
    // msg = sigmoid(gate) * silu(core) * mask_j ; reduce over M (=j rows)
    float part = 0.f;
    #pragma unroll
    for (int r = 0; r < 8; ++r) {
      float g = accg[r], c = accc[r];
      part += sigm(g) * (c * sigm(c)) * mj[r];
    }
    part += __shfl_xor(part, 16, 32);      // fold the two M half-tiles
    if (lane < 16) sagg[jt * HD + t * 16 + m] = part;
  }
  __syncthreads();

  // cross-wave (cross j-tile) reduction + residual + conditioning + mask_i
  if (threadIdx.x < HD) {
    int c = threadIdx.x;
    float s = 0.f;
    #pragma unroll
    for (int q = 0; q < 8; ++q) s += sagg[q * HD + c];
    size_t off = ((size_t)b * NA + i) * HD + c;
    float hv = (hf_in[off] + s + cnd[b * HD + c]) * smask[i];
    hf_out[off]  = hv;
    h16_out[off] = (_Float16)hv;
  }
}

// ------------------------- output predictors -------------------------------

__global__ void dd_pred_hidden_kernel(const float* __restrict__ h,
                                      const float* __restrict__ Wc1,
                                      const float* __restrict__ bc1,
                                      float* __restrict__ tmp) {
  int idx = blockIdx.x * blockDim.x + threadIdx.x;       // row*HD + c
  if (idx >= NB * NA * HD) return;
  int row = idx >> 7, c = idx & 127;
  const float* hr = h + (size_t)row * HD;
  float acc = bc1[c];
  #pragma unroll 8
  for (int k = 0; k < HD; ++k) acc += hr[k] * Wc1[k * HD + c];
  tmp[idx] = acc * sigm(acc);                            // silu
}

__global__ void dd_pred_noise_kernel(const float* __restrict__ tmp,
                                     const float* __restrict__ Wc2,
                                     const float* __restrict__ bc2,
                                     float* __restrict__ out) {
  int idx = blockIdx.x * blockDim.x + threadIdx.x;       // row*3 + c
  if (idx >= NB * NA * 3) return;
  int row = idx / 3, c = idx % 3;
  const float* tr = tmp + (size_t)row * HD;
  float acc = bc2[c];
  #pragma unroll 8
  for (int k = 0; k < HD; ++k) acc += tr[k] * Wc2[k * 3 + c];
  out[idx] = acc;
}

__global__ void dd_pred_types_kernel(const float* __restrict__ h,
                                     const float* __restrict__ Wt,
                                     const float* __restrict__ bt,
                                     float* __restrict__ out) {
  int idx = blockIdx.x * blockDim.x + threadIdx.x;       // row*VOC + c
  if (idx >= NB * NA * VOC) return;
  int row = idx / VOC, c = idx % VOC;
  const float* hr = h + (size_t)row * HD;
  float acc = bt[c];
  #pragma unroll 8
  for (int k = 0; k < HD; ++k) acc += hr[k] * Wt[k * VOC + c];
  out[idx] = acc;
}

// ------------------------- launch ------------------------------------------

extern "C" void kernel_launch(void* const* d_in, const int* in_sizes, int n_in,
                              void* d_out, int out_size, void* d_ws, size_t ws_size,
                              hipStream_t stream) {
  const int*   atom_types = (const int*)  d_in[0];
  const float* frac       = (const float*)d_in[1];
  const float* lat        = (const float*)d_in[2];
  const unsigned char* mask = (const unsigned char*)d_in[3];
  const int*   tstep      = (const int*)  d_in[4];
  const float* z          = (const float*)d_in[5];
  const float* atom_emb   = (const float*)d_in[6];
  const float* Wlat       = (const float*)d_in[7];
  const float* blat       = (const float*)d_in[8];
  const float* edge_W     = (const float*)d_in[9];
  const float* edge_b     = (const float*)d_in[10];
  const float* cond_W     = (const float*)d_in[11];
  const float* cond_b     = (const float*)d_in[12];
  const float* Wc1        = (const float*)d_in[13];
  const float* bc1        = (const float*)d_in[14];
  const float* Wc2        = (const float*)d_in[15];
  const float* bc2        = (const float*)d_in[16];
  const float* Wt         = (const float*)d_in[17];
  const float* bt         = (const float*)d_in[18];

  char* ws = (char*)d_ws;
  float*     cart  = (float*)    (ws + OFF_CART);
  float*     cond  = (float*)    (ws + OFF_COND);
  float*     cnd   = (float*)    (ws + OFF_CND);
  _Float16*  wz16  = (_Float16*) (ws + OFF_WSWZ);
  _Float16*  h16[2] = { (_Float16*)(ws + OFF_H16A), (_Float16*)(ws + OFF_H16B) };
  float*     hf[2]  = { (float*)   (ws + OFF_HFA),  (float*)   (ws + OFF_HFB)  };
  float*     tmp   = (float*)    (ws + OFF_TMP);
  _Float16*  rbf16 = (_Float16*) (ws + OFF_RBF);
  float*     out   = (float*)d_out;

  dd_cart_kernel<<<(NB*NA + 255)/256, 256, 0, stream>>>(frac, lat, cart);
  dd_h0_kernel<<<(NB*NA*HD + 255)/256, 256, 0, stream>>>(atom_types, atom_emb, hf[0], h16[0]);
  dd_wswz_kernel<<<(NL*KS*16*32 + 255)/256, 256, 0, stream>>>(edge_W, wz16);
  dd_cond_kernel<<<(NB*256 + 255)/256, 256, 0, stream>>>(tstep, z, Wlat, blat, cond);
  dd_rbf_kernel<<<(NB*NA*NA + 255)/256, 256, 0, stream>>>(cart, rbf16);

  int ping = 0;
  for (int l = 0; l < NL; ++l) {
    dd_cnd_kernel<<<(NB*HD + 255)/256, 256, 0, stream>>>(
        cond, cond_W + (size_t)l * 256 * HD, cond_b + (size_t)l * HD, cnd);
    dd_edge_kernel<<<dim3(NA, NB), 256, 0, stream>>>(
        h16[ping], hf[ping], rbf16,
        (const uint4*)(wz16 + (size_t)l * KS * 16 * 32 * 16),
        edge_b + (size_t)l * 256, cnd, mask,
        hf[1 - ping], h16[1 - ping]);
    ping = 1 - ping;
  }

  dd_pred_hidden_kernel<<<(NB*NA*HD + 255)/256, 256, 0, stream>>>(hf[ping], Wc1, bc1, tmp);
  dd_pred_noise_kernel<<<(NB*NA*3 + 255)/256, 256, 0, stream>>>(tmp, Wc2, bc2, out);
  dd_pred_types_kernel<<<(NB*NA*VOC + 255)/256, 256, 0, stream>>>(hf[ping], Wt, bt, out + NB*NA*3);
}
